// SparseAttentionBlock_19267223289859
// MI455X (gfx1250) — matrix operand order repaired and checked by reading, not compiled, not run on hardware
//
#include <hip/hip_runtime.h>
#include <hip/hip_bf16.h>

// ---------------------------------------------------------------------------
// Types for WMMA / TDM
// ---------------------------------------------------------------------------
typedef __attribute__((ext_vector_type(16))) _Float16 v16h;
typedef __attribute__((ext_vector_type(8)))  float    v8f;
typedef __attribute__((ext_vector_type(4)))  unsigned u32x4;
typedef __attribute__((ext_vector_type(8)))  unsigned u32x8;

#define NUM_HEADS 8
#define HEAD_DIM  64
#define EMBED     512
#define POOLK     4
#define TOPK      8
#define SCALING   0.125f   // 64^-0.5

__device__ __forceinline__ void wait_tensorcnt0() {
#if __has_builtin(__builtin_amdgcn_s_wait_tensorcnt)
    __builtin_amdgcn_s_wait_tensorcnt(0);
#else
    asm volatile("s_wait_tensorcnt 0x0" ::: "memory");
#endif
}

__device__ __forceinline__ void wait_asynccnt0() {
#if __has_builtin(__builtin_amdgcn_s_wait_asynccnt)
    __builtin_amdgcn_s_wait_asynccnt(0);
#else
    asm volatile("s_wait_asynccnt 0x0" ::: "memory");
#endif
}

// ---------------------------------------------------------------------------
// f32 -> f16 conversion
// ---------------------------------------------------------------------------
__global__ __launch_bounds__(256) void cvt_f32_f16(const float* __restrict__ src,
                                                   _Float16* __restrict__ dst, int n) {
    int i = blockIdx.x * 256 + threadIdx.x;
    if (i < n) dst[i] = (_Float16)src[i];
}

__global__ __launch_bounds__(256) void zero_u32(unsigned* __restrict__ p, int n) {
    int i = blockIdx.x * 256 + threadIdx.x;
    if (i < n) p[i] = 0u;
}

// ---------------------------------------------------------------------------
// Tiled WMMA GEMM: C[MxN] = A[MxK] * B[KxN], f16 inputs, f32 accumulate.
// Block tile 64x128, 8 waves, each wave a 32x32 subtile (2x2 WMMA tiles).
// A tile (64x32, row-major, stride K) is DMA'd into LDS by wave 0 via the
// Tensor Data Mover (tensor_load_to_lds, TENSORcnt), double-buffered so the
// DMA of tile k+1 overlaps the WMMA work on tile k. B tile is register-staged
// and stored TRANSPOSED [n][k] so each B fragment is two contiguous 16B DS
// loads per lane. One barrier per K-step.
// ---------------------------------------------------------------------------
#define BM 64
#define BN 128
#define BK 32

template <typename OutT>
__global__ __launch_bounds__(256) void gemm16(const _Float16* __restrict__ A,
                                              const _Float16* __restrict__ B,
                                              OutT* __restrict__ C,
                                              int M, int N, int K) {
    __shared__ _Float16 As[2][BM * BK];   // [m][k]
    __shared__ _Float16 Bs[2][BN * BK];   // [n][k] (transposed)

    const int tid   = threadIdx.x;
    const int lane  = tid & 31;
    const int wave  = tid >> 5;
    const int wm    = wave >> 2;          // 0..1
    const int wn    = wave & 3;           // 0..3
    const int bm0   = blockIdx.y * BM;
    const int bn0   = blockIdx.x * BN;
    const int l15   = lane & 15;
    const int lhalf = lane >> 4;
    const int nt    = K / BK;

    // --- TDM: DMA one 64x32 f16 tile of A into LDS buffer `buf` (wave0 only)
    auto issue_tdm = [&](int kt, int buf) {
        unsigned lds = (unsigned)(size_t)(&As[buf][0]);   // low 32 bits = LDS offset
        unsigned long long ga =
            (unsigned long long)(size_t)(A + (size_t)bm0 * K + (size_t)kt * BK);
        u32x4 g0;
        g0[0] = 1u;                                        // count=1, user mode
        g0[1] = lds;                                       // lds_addr
        g0[2] = (unsigned)(ga & 0xffffffffu);              // global_addr[31:0]
        g0[3] = (unsigned)((ga >> 32) & 0x1ffffffu) | (2u << 30);  // addr[56:32] | type=2
        u32x8 g1;
        g1[0] = 1u << 16;                                  // data_size = 2 bytes
        g1[1] = ((unsigned)K & 0xffffu) << 16;             // tensor_dim0[15:0] @ bits63:48
        g1[2] = (((unsigned)K >> 16) & 0xffffu)            // tensor_dim0[31:16]
              | (((unsigned)M & 0xffffu) << 16);           // tensor_dim1[15:0]
        g1[3] = (((unsigned)M >> 16) & 0xffffu)            // tensor_dim1[31:16]
              | ((unsigned)BK << 16);                      // tile_dim0 = 32
        g1[4] = (unsigned)BM;                              // tile_dim1 = 64
        g1[5] = (unsigned)K;                               // tensor_dim0_stride[31:0]
        g1[6] = 0u;
        g1[7] = 0u;
        asm volatile("tensor_load_to_lds %0, %1" :: "s"(g0), "s"(g1) : "memory");
    };

    // --- B tile: global -> regs (issued early) and regs -> LDS transposed
    uint4 breg[2];
    auto loadB = [&](int kt) {
        #pragma unroll
        for (int i = 0; i < 2; ++i) {
            int chunk = tid + i * 256;    // 0..511
            int kk    = chunk >> 4;       // 0..31
            int nn    = (chunk & 15) * 8; // 0..120
            breg[i] = *reinterpret_cast<const uint4*>(
                B + (size_t)(kt * BK + kk) * N + bn0 + nn);
        }
    };
    auto storeB = [&](int buf) {
        #pragma unroll
        for (int i = 0; i < 2; ++i) {
            int chunk = tid + i * 256;
            int kk    = chunk >> 4;
            int nn    = (chunk & 15) * 8;
            const _Float16* dp = reinterpret_cast<const _Float16*>(&breg[i]);
            #pragma unroll
            for (int j = 0; j < 8; ++j) Bs[buf][(nn + j) * BK + kk] = dp[j];
        }
    };

    v8f acc[2][2] = {};

    // --- prologue: fill buffer 0
    if (tid < 32) issue_tdm(0, 0);
    loadB(0);
    storeB(0);
    if (tid < 32) wait_tensorcnt0();
    __syncthreads();

    for (int it = 0; it < nt; ++it) {
        const int cur = it & 1;
        const int nxt = cur ^ 1;
        const bool has_next = (it + 1 < nt);

        if (has_next) {
            if (tid < 32) issue_tdm(it + 1, nxt);   // DMA overlaps compute below
            loadB(it + 1);                          // VMEM latency overlaps compute
        }

        // ---- fragments in contiguous k order (k = 16*lhalf + s) for A and B (from buffer `cur`)
        union AU { uint32_t u[8]; v16h v; };
        union BU { uint4 q[2];   v16h v; };
        AU afr[2];
        BU bfr[2];
        #pragma unroll
        for (int ta = 0; ta < 2; ++ta) {
            int m = wm * 32 + ta * 16 + l15;
            #pragma unroll
            for (int p = 0; p < 8; ++p) {
                int koff = lhalf * 16 + 2 * p;
                afr[ta].u[p] =
                    *reinterpret_cast<const uint32_t*>(&As[cur][m * BK + koff]);
            }
        }
        #pragma unroll
        for (int tb = 0; tb < 2; ++tb) {
            int n  = wn * 32 + tb * 16 + l15;
            int kb = lhalf * 16;
            const uint4* bp = reinterpret_cast<const uint4*>(&Bs[cur][n * BK + kb]);
            bfr[tb].q[0] = bp[0];
            bfr[tb].q[1] = bp[1];
        }

        #pragma unroll
        for (int ta = 0; ta < 2; ++ta)
            #pragma unroll
            for (int tb = 0; tb < 2; ++tb)
                acc[ta][tb] = __builtin_amdgcn_wmma_f32_16x16x32_f16(
                    false, afr[ta].v, false, bfr[tb].v,
                    (short)0, acc[ta][tb], false, false);

        if (has_next) storeB(nxt);          // write next buffer (no reader yet)
        if (tid < 32) wait_tensorcnt0();    // next A tile landed
        __syncthreads();                    // one barrier per iteration
    }

    // ---- store D (C/D layout: lane n = l15, rows j + 8*lhalf)
    #pragma unroll
    for (int ta = 0; ta < 2; ++ta)
        #pragma unroll
        for (int tb = 0; tb < 2; ++tb)
            #pragma unroll
            for (int j = 0; j < 8; ++j) {
                int m = bm0 + wm * 32 + ta * 16 + j + lhalf * 8;
                int n = bn0 + wn * 32 + tb * 16 + l15;
                C[(size_t)m * N + n] = (OutT)acc[ta][tb][j];
            }
}

// ---------------------------------------------------------------------------
// Chunk-mean pooling: (b, s, E) f16 -> (b, nc, E) f16, mean over 4 rows
// ---------------------------------------------------------------------------
__global__ __launch_bounds__(256) void pool_kernel(const _Float16* __restrict__ src,
                                                   _Float16* __restrict__ dst,
                                                   int S, int NC, int total) {
    int idx = blockIdx.x * 256 + threadIdx.x;
    if (idx >= total) return;
    int b   = idx / (NC * EMBED);
    int rem = idx % (NC * EMBED);
    int c   = rem / EMBED;
    int e   = rem % EMBED;
    const _Float16* p = src + ((size_t)(b * S + c * POOLK) * EMBED) + e;
    float a = 0.f;
    #pragma unroll
    for (int r = 0; r < POOLK; ++r) a += (float)p[(size_t)r * EMBED];
    dst[idx] = (_Float16)(a * 0.25f);
}

// ---------------------------------------------------------------------------
// Routing: one wave per 16(query) x 16(chunk) WMMA score tile for one (b,h);
// K = 64 = two v_wmma_f32_16x16x32_f16; max-reduce + monotone-encoded
// atomicMax into per-(b,chunk) table.
// ---------------------------------------------------------------------------
__device__ __forceinline__ unsigned enc_f32(float f) {
    unsigned u = __float_as_uint(f);
    return (u & 0x80000000u) ? ~u : (u | 0x80000000u);
}

__global__ __launch_bounds__(32) void route_kernel(const _Float16* __restrict__ Q,
                                                   const _Float16* __restrict__ PK,
                                                   unsigned* __restrict__ route,
                                                   int S, int NC) {
    const int b  = blockIdx.z >> 3;
    const int h  = blockIdx.z & 7;
    const int s0 = blockIdx.y * 16;
    const int c0 = blockIdx.x * 16;
    const int lane  = threadIdx.x;
    const int l15   = lane & 15;
    const int lhalf = lane >> 4;

    union AU { uint32_t u[8]; v16h v; };
    union BU { uint4 q[2];   v16h v; };

    const _Float16* qbase = Q  + ((size_t)(b * S  + s0) * EMBED) + h * HEAD_DIM;
    const _Float16* kbase = PK + ((size_t)(b * NC + c0) * EMBED) + h * HEAD_DIM;

    v8f acc = {};
    #pragma unroll
    for (int kk = 0; kk < 2; ++kk) {
        AU a; BU bb;
        #pragma unroll
        for (int p = 0; p < 8; ++p) {
            int koff = kk * 32 + lhalf * 16 + 2 * p;
            a.u[p] = *reinterpret_cast<const uint32_t*>(qbase + (size_t)l15 * EMBED + koff);
        }
        int kb = kk * 32 + lhalf * 16;
        const uint4* bp = reinterpret_cast<const uint4*>(kbase + (size_t)l15 * EMBED + kb);
        bb.q[0] = bp[0];
        bb.q[1] = bp[1];
        acc = __builtin_amdgcn_wmma_f32_16x16x32_f16(false, a.v, false, bb.v,
                                                     (short)0, acc, false, false);
    }

    float mx = acc[0];
    #pragma unroll
    for (int j = 1; j < 8; ++j) mx = fmaxf(mx, acc[j]);
    mx *= SCALING;
    mx = fmaxf(mx, __shfl_xor(mx, 16));
    if (lane < 16) atomicMax(&route[b * NC + c0 + lane], enc_f32(mx));
}

// ---------------------------------------------------------------------------
// Top-8 per batch; LDS table filled with async-to-LDS loads (ASYNCcnt).
// ---------------------------------------------------------------------------
__global__ __launch_bounds__(256) void topk_kernel(const unsigned* __restrict__ route,
                                                   int* __restrict__ topidx, int NC) {
    __shared__ unsigned vals[1024];
    __shared__ unsigned rv[256];
    __shared__ int      ri[256];
    const int b = blockIdx.x, t = threadIdx.x;

    // async copy: each thread DMAs 16B straight into LDS (no VGPR round trip)
    {
        unsigned lds = (unsigned)(size_t)(&vals[t * 4]);
        unsigned long long ga =
            (unsigned long long)(size_t)(route + (size_t)b * NC + t * 4);
        asm volatile("global_load_async_to_lds_b128 %0, %1, off"
                     :: "v"(lds), "v"(ga) : "memory");
    }
    wait_asynccnt0();
    __syncthreads();

    for (int sel = 0; sel < TOPK; ++sel) {
        unsigned bestv = 0u; int besti = NC;
        for (int i = t; i < NC; i += 256) {
            unsigned v = vals[i];
            if (v > bestv || (v == bestv && i < besti)) { bestv = v; besti = i; }
        }
        rv[t] = bestv; ri[t] = besti;
        __syncthreads();
        for (int off = 128; off > 0; off >>= 1) {
            if (t < off) {
                if (rv[t + off] > rv[t] || (rv[t + off] == rv[t] && ri[t + off] < ri[t])) {
                    rv[t] = rv[t + off]; ri[t] = ri[t + off];
                }
            }
            __syncthreads();
        }
        if (t == 0) { topidx[b * TOPK + sel] = ri[0]; vals[ri[0]] = 0u; }
        __syncthreads();
    }
}

// ---------------------------------------------------------------------------
// Sparse attention over the 8 selected chunks: 1 thread per (b,h,s) row.
// ---------------------------------------------------------------------------
__global__ __launch_bounds__(256) void attn_kernel(const _Float16* __restrict__ Q,
                                                   const _Float16* __restrict__ PK,
                                                   const _Float16* __restrict__ PV,
                                                   const int* __restrict__ topidx,
                                                   _Float16* __restrict__ Out,
                                                   int S, int NC) {
    int idx = blockIdx.x * 256 + threadIdx.x;
    int b   = idx / (NUM_HEADS * S);
    int rem = idx % (NUM_HEADS * S);
    int h   = rem / S;
    int s   = rem % S;

    const _Float16* q = Q + ((size_t)(b * S + s) * EMBED) + h * HEAD_DIM;
    int cs[TOPK];
    #pragma unroll
    for (int j = 0; j < TOPK; ++j) cs[j] = topidx[b * TOPK + j];

    float sc[TOPK];
    #pragma unroll
    for (int j = 0; j < TOPK; ++j) {
        const _Float16* kp = PK + ((size_t)(b * NC + cs[j]) * EMBED) + h * HEAD_DIM;
        float a = 0.f;
        #pragma unroll 8
        for (int d = 0; d < HEAD_DIM; ++d) a += (float)q[d] * (float)kp[d];
        sc[j] = a * SCALING;
    }
    float m = sc[0];
    #pragma unroll
    for (int j = 1; j < TOPK; ++j) m = fmaxf(m, sc[j]);
    float den = 0.f;
    #pragma unroll
    for (int j = 0; j < TOPK; ++j) { sc[j] = __expf(sc[j] - m); den += sc[j]; }
    float inv = 1.f / den;
    #pragma unroll
    for (int j = 0; j < TOPK; ++j) sc[j] *= inv;

    _Float16* op = Out + ((size_t)(b * S + s) * EMBED) + h * HEAD_DIM;
    #pragma unroll 8
    for (int d = 0; d < HEAD_DIM; ++d) {
        float o = 0.f;
        #pragma unroll
        for (int j = 0; j < TOPK; ++j) {
            const _Float16* vp = PV + ((size_t)(b * NC + cs[j]) * EMBED) + h * HEAD_DIM;
            o += sc[j] * (float)vp[d];
        }
        op[d] = (_Float16)o;
    }
}

// ---------------------------------------------------------------------------
// Launch
// ---------------------------------------------------------------------------
extern "C" void kernel_launch(void* const* d_in, const int* in_sizes, int n_in,
                              void* d_out, int out_size, void* d_ws, size_t ws_size,
                              hipStream_t stream) {
    const float* X  = (const float*)d_in[0];
    const float* Wq = (const float*)d_in[1];
    const float* Wk = (const float*)d_in[2];
    const float* Wv = (const float*)d_in[3];
    const float* Wo = (const float*)d_in[4];

    const int B = 4, S = 4096, NC = S / POOLK;   // 1024
    const int M = B * S;                          // 16384
    const int E = EMBED;                          // 512

    char* ws = (char*)d_ws;
    size_t off = 0;
    auto take = [&](size_t bytes) -> void* {
        void* p = ws + off;
        off = (off + bytes + 255) & ~(size_t)255;
        return p;
    };

    _Float16* Xh   = (_Float16*)take((size_t)M * E * 2);   // reused as attn out
    _Float16* Wqh  = (_Float16*)take((size_t)E * E * 2);
    _Float16* Wkh  = (_Float16*)take((size_t)E * E * 2);
    _Float16* Wvh  = (_Float16*)take((size_t)E * E * 2);
    _Float16* Woh  = (_Float16*)take((size_t)E * E * 2);
    _Float16* Qh   = (_Float16*)take((size_t)M * E * 2);
    _Float16* KVt  = (_Float16*)take((size_t)M * E * 2);   // K then V projection temp
    _Float16* PKh  = (_Float16*)take((size_t)B * NC * E * 2);
    _Float16* PVh  = (_Float16*)take((size_t)B * NC * E * 2);
    unsigned* Rt   = (unsigned*)take((size_t)B * NC * 4);
    int*      Tix  = (int*)take((size_t)B * TOPK * 4);

    cvt_f32_f16<<<(M * E + 255) / 256, 256, 0, stream>>>(X, Xh, M * E);
    cvt_f32_f16<<<(E * E + 255) / 256, 256, 0, stream>>>(Wq, Wqh, E * E);
    cvt_f32_f16<<<(E * E + 255) / 256, 256, 0, stream>>>(Wk, Wkh, E * E);
    cvt_f32_f16<<<(E * E + 255) / 256, 256, 0, stream>>>(Wv, Wvh, E * E);
    cvt_f32_f16<<<(E * E + 255) / 256, 256, 0, stream>>>(Wo, Woh, E * E);

    dim3 ggrid(E / BN, M / BM);   // (4, 256)

    gemm16<_Float16><<<ggrid, 256, 0, stream>>>(Xh, Wkh, KVt, M, E, E);
    pool_kernel<<<(B * NC * E + 255) / 256, 256, 0, stream>>>(KVt, PKh, S, NC, B * NC * E);
    gemm16<_Float16><<<ggrid, 256, 0, stream>>>(Xh, Wvh, KVt, M, E, E);
    pool_kernel<<<(B * NC * E + 255) / 256, 256, 0, stream>>>(KVt, PVh, S, NC, B * NC * E);
    gemm16<_Float16><<<ggrid, 256, 0, stream>>>(Xh, Wqh, Qh, M, E, E);

    zero_u32<<<(B * NC + 255) / 256, 256, 0, stream>>>(Rt, B * NC);
    route_kernel<<<dim3(NC / 16, S / 16, B * NUM_HEADS), 32, 0, stream>>>(Qh, PKh, Rt, S, NC);
    topk_kernel<<<B, 256, 0, stream>>>(Rt, Tix, NC);

    attn_kernel<<<(B * NUM_HEADS * S) / 256, 256, 0, stream>>>(Qh, PKh, PVh, Tix, Xh, S, NC);

    gemm16<float><<<ggrid, 256, 0, stream>>>(Xh, Woh, (float*)d_out, M, E, E);
}
